// SmilesTransformer_46720654246237
// MI455X (gfx1250) — compile-verified, matrix operand order checked by reading, not run on hardware
//
#include <hip/hip_runtime.h>

// ---------------- CDNA5 WMMA types / helpers ----------------
typedef __attribute__((ext_vector_type(16))) __bf16 v16bf;
typedef __attribute__((ext_vector_type(8)))  float  v8f;

static __device__ __forceinline__ unsigned short f2bf(float f) {
    unsigned int u = __builtin_bit_cast(unsigned int, f);
    u = (u + 0x7fffu + ((u >> 16) & 1u)) >> 16;
    return (unsigned short)u;
}
static __device__ __forceinline__ __bf16 bfc(unsigned short u) {
    return __builtin_bit_cast(__bf16, u);
}
// ds_swizzle xor-lane exchange (wave32-safe, no wave64 assumptions)
#define XORF(v, m) (__builtin_bit_cast(float, __builtin_amdgcn_ds_swizzle( \
    __builtin_bit_cast(int, (float)(v)), (0x1f | ((m) << 10)))))

// async global->LDS copy (CDNA5): copies 16B; INST_OFFSET applies to both addresses
#define ASYNC_B128(ldsoff, gaddr) \
    asm volatile("global_load_async_to_lds_b128 %0, %1, off" \
                 :: "v"(ldsoff), "v"(gaddr) : "memory")
#define ASYNC_B128_OFS(ldsoff, gaddr, ofs) \
    asm volatile("global_load_async_to_lds_b128 %0, %1, off offset:" #ofs \
                 :: "v"(ldsoff), "v"(gaddr) : "memory")
#define WAIT_ASYNC(n) asm volatile("s_wait_asynccnt " #n ::: "memory")

#define TOKENS 4096   // B*L
#define LL     2048
#define DD     256
#define HH_N   8
#define DH     32
#define TD     768    // 3*D
#define FFN    1024
#define VPAD   128    // padded vocab rows for LM-head B-tile

// ---------------- convert kernels ----------------
// Transposed weight conversion: out[l][n][k] = bf16(in[l][k][n]); total = NL*N*K
__global__ __launch_bounds__(256) void k_wT(const float* __restrict__ in,
                                            unsigned short* __restrict__ out,
                                            int K, int N, int total) {
    for (int i = blockIdx.x * 256 + threadIdx.x; i < total; i += gridDim.x * 256) {
        int l   = i / (N * K);
        int rem = i - l * (N * K);
        int n   = rem / K;
        int k   = rem - n * K;
        out[i] = f2bf(in[(size_t)l * K * N + (size_t)k * N + n]);
    }
}
// LM head Bt = emb rows (V x D), zero-padded to VPAD rows
__global__ __launch_bounds__(256) void k_lmhead(const float* __restrict__ emb,
                                                unsigned short* __restrict__ out) {
    for (int i = blockIdx.x * 256 + threadIdx.x; i < VPAD * DD; i += gridDim.x * 256) {
        int v = i >> 8, k = i & 255;
        out[i] = (v < 100) ? f2bf(emb[(size_t)v * DD + k]) : (unsigned short)0;
    }
}

// ---------------- embedding + positional encoding ----------------
__global__ __launch_bounds__(256) void k_embed(const int* __restrict__ x,
                                               const float* __restrict__ emb,
                                               const float* __restrict__ pe,
                                               float* __restrict__ h) {
    int t = blockIdx.x, c = threadIdx.x;
    int tok = x[t];
    int l = t & (LL - 1);
    h[(size_t)t * DD + c] = emb[(size_t)tok * DD + c] + pe[(size_t)l * DD + c];
}

// ---------------- LayerNorm -> bf16 (8 rows/block, 1 wave/row) ----------------
__global__ __launch_bounds__(256) void k_ln(const float* __restrict__ in,
                                            const float* __restrict__ w,
                                            const float* __restrict__ b,
                                            unsigned short* __restrict__ out) {
    int wave = threadIdx.x >> 5, lane = threadIdx.x & 31;
    int row  = blockIdx.x * 8 + wave;
    const float* p = in + (size_t)row * DD + lane * 8;
    float4 v0 = *(const float4*)p;
    float4 v1 = *(const float4*)(p + 4);
    float vals[8] = {v0.x, v0.y, v0.z, v0.w, v1.x, v1.y, v1.z, v1.w};
    float s = 0.f, q = 0.f;
#pragma unroll
    for (int i = 0; i < 8; ++i) { s += vals[i]; q += vals[i] * vals[i]; }
    s += XORF(s, 1);  q += XORF(q, 1);
    s += XORF(s, 2);  q += XORF(q, 2);
    s += XORF(s, 4);  q += XORF(q, 4);
    s += XORF(s, 8);  q += XORF(q, 8);
    s += XORF(s, 16); q += XORF(q, 16);
    float mean = s * (1.0f / DD);
    float var  = q * (1.0f / DD) - mean * mean;
    float rs   = rsqrtf(var + 1e-5f);
#pragma unroll
    for (int i = 0; i < 8; ++i) {
        int c = lane * 8 + i;
        out[(size_t)row * DD + c] = f2bf((vals[i] - mean) * rs * w[c] + b[c]);
    }
}

// ---------------- WMMA GEMM: C[M,N](f32) = A[M,K](bf16) @ Bt[N,K](bf16)^T ----------------
// Bt is N-major (pre-transposed). Double-buffered LDS filled by async
// global->LDS copies (ASYNCcnt); copy of chunk i+1 overlaps WMMA on chunk i.
// Pipeline is peeled/unrolled x2 so buffer indices are compile-time constants.
typedef unsigned short lds_tile_t[2][64][40];

template <int BUF>
static __device__ __forceinline__ void gemm_mma(const unsigned short (&As)[2][64][40],
                                                const unsigned short (&Bs)[2][64][40],
                                                int mw, int nw, int hl, int mr,
                                                v8f& acc0, v8f& acc1) {
    v16bf fa, fb0, fb1;
#pragma unroll
    for (int p = 0; p < 8; ++p) {
        int kk = ((p >> 2) << 4) + hl * 8 + 2 * (p & 3);
        unsigned int ua = *(const unsigned int*)&As[BUF][mw * 16 + mr][kk];
        unsigned int u0 = *(const unsigned int*)&Bs[BUF][nw * 32 + mr][kk];
        unsigned int u1 = *(const unsigned int*)&Bs[BUF][nw * 32 + 16 + mr][kk];
        fa[2 * p] = bfc((unsigned short)ua);  fa[2 * p + 1] = bfc((unsigned short)(ua >> 16));
        fb0[2 * p] = bfc((unsigned short)u0); fb0[2 * p + 1] = bfc((unsigned short)(u0 >> 16));
        fb1[2 * p] = bfc((unsigned short)u1); fb1[2 * p + 1] = bfc((unsigned short)(u1 >> 16));
    }
    acc0 = __builtin_amdgcn_wmma_f32_16x16x32_bf16(false, fa, false, fb0, (short)0, acc0, false, false);
    acc1 = __builtin_amdgcn_wmma_f32_16x16x32_bf16(false, fa, false, fb1, (short)0, acc1, false, false);
}

__global__ __launch_bounds__(256) void k_gemm(const unsigned short* __restrict__ A,
                                              const unsigned short* __restrict__ Bt,
                                              int M, int N, int K,
                                              const float* __restrict__ bias,
                                              const float* __restrict__ res,
                                              float* __restrict__ outF,
                                              unsigned short* __restrict__ outB,
                                              int dogelu) {
    __shared__ unsigned short As[2][64][40];   // [buf][m][k], row stride 80B
    __shared__ unsigned short Bs[2][64][40];   // [buf][n][k]
    const int tid  = threadIdx.x;
    const int lane = tid & 31, wave = tid >> 5;
    const int mw = wave >> 1, nw = wave & 1;
    const int hl = lane >> 4, mr = lane & 15;
    const int m0 = blockIdx.y * 64, n0 = blockIdx.x * 64;
    const int r  = tid >> 2, cg = (tid & 3) * 8;

    const unsigned short* aSrc = A  + (size_t)(m0 + r) * K + cg;
    const unsigned short* bSrc = Bt + (size_t)(n0 + r) * K + cg;
    const unsigned aDst0 = (unsigned)(size_t)&As[0][r][cg];
    const unsigned aDst1 = (unsigned)(size_t)&As[1][r][cg];
    const unsigned bDst0 = (unsigned)(size_t)&Bs[0][r][cg];
    const unsigned bDst1 = (unsigned)(size_t)&Bs[1][r][cg];

    // prologue: chunk 0 -> buf 0
    ASYNC_B128(aDst0, (unsigned long long)aSrc);
    ASYNC_B128(bDst0, (unsigned long long)bSrc);

    v8f acc0 = {}, acc1 = {};
    const int nk = K >> 5;   // always even (K = 256 or 1024)

    // steady state: pairs of steps with constant buffer indices
    for (int i = 0; i + 2 < nk; i += 2) {
        // step i (buf 0): prefetch chunk i+1 -> buf 1
        ASYNC_B128(aDst1, (unsigned long long)(aSrc + (i + 1) * 32));
        ASYNC_B128(bDst1, (unsigned long long)(bSrc + (i + 1) * 32));
        WAIT_ASYNC(0x2);
        __syncthreads();
        gemm_mma<0>(As, Bs, mw, nw, hl, mr, acc0, acc1);
        __syncthreads();
        // step i+1 (buf 1): prefetch chunk i+2 -> buf 0
        ASYNC_B128(aDst0, (unsigned long long)(aSrc + (i + 2) * 32));
        ASYNC_B128(bDst0, (unsigned long long)(bSrc + (i + 2) * 32));
        WAIT_ASYNC(0x2);
        __syncthreads();
        gemm_mma<1>(As, Bs, mw, nw, hl, mr, acc0, acc1);
        __syncthreads();
    }
    // epilogue steps nk-2 (buf 0, prefetch last chunk) and nk-1 (buf 1, drain)
    ASYNC_B128(aDst1, (unsigned long long)(aSrc + (nk - 1) * 32));
    ASYNC_B128(bDst1, (unsigned long long)(bSrc + (nk - 1) * 32));
    WAIT_ASYNC(0x2);
    __syncthreads();
    gemm_mma<0>(As, Bs, mw, nw, hl, mr, acc0, acc1);
    __syncthreads();
    WAIT_ASYNC(0x0);
    __syncthreads();
    gemm_mma<1>(As, Bs, mw, nw, hl, mr, acc0, acc1);

#pragma unroll
    for (int sub = 0; sub < 2; ++sub) {
        v8f acc = sub ? acc1 : acc0;
        int n = n0 + nw * 32 + sub * 16 + mr;
        if (n < N) {
#pragma unroll
            for (int rr = 0; rr < 8; ++rr) {
                int m = m0 + mw * 16 + 8 * hl + rr;
                float v = acc[rr];
                if (bias)   v += bias[n];
                if (dogelu) v = 0.5f * v * (1.0f + erff(v * 0.7071067811865476f));
                if (res)    v += res[(size_t)m * N + n];
                if (outF)   outF[(size_t)m * N + n] = v;
                if (outB)   outB[(size_t)m * N + n] = f2bf(v);
            }
        }
    }
}

// ---------------- Flash attention: 1 wave per (b, head, 16-query tile) ----------------
// qkv bf16 [4096, 768] (q|k|v), o bf16 [4096, 256]
__global__ __launch_bounds__(32) void k_attn(const unsigned short* __restrict__ qkv,
                                             const int* __restrict__ x,
                                             unsigned short* __restrict__ o) {
    const int lane = threadIdx.x;
    const int hl = lane >> 4, mr = lane & 15;
    int bid = blockIdx.x;
    int qt = bid & 127;
    int hh = (bid >> 7) & 7;
    int b  = bid >> 10;
    int q0 = qt * 16;
    int rowbase = b * LL;

    __shared__ float row_m[16], row_l[16], row_sc[16], kmask[32];
    __shared__ unsigned short p_lds[16][36];   // [query][key]
    __shared__ unsigned short v_lds[32][40];   // [key][dh], row stride 80B

    if (lane < 16) { row_m[lane] = -1e30f; row_l[lane] = 0.0f; }
    __syncthreads();

    // Q fragment (B operand of S^T = K @ Q^T; layout identical to A mapping)
    v16bf fq;
    {
        const unsigned short* qp = qkv + (size_t)(rowbase + q0 + mr) * TD + hh * DH;
#pragma unroll
        for (int p = 0; p < 8; ++p) {
            int kk = ((p >> 2) << 4) + hl * 8 + 2 * (p & 3);
            unsigned int u = *(const unsigned int*)(qp + kk);
            fq[2 * p] = bfc((unsigned short)u); fq[2 * p + 1] = bfc((unsigned short)(u >> 16));
        }
    }

    const unsigned vDst = (unsigned)(size_t)&v_lds[lane][0];

    v8f o0 = {}, o1 = {};
    const float scal = 0.17677669529663687f;   // 1/sqrt(32)
    int iters = (q0 + 16 + 31) >> 5;

    for (int it = 0; it < iters; ++it) {
        int key0 = it * 32;
        // async-stage the V chunk (one key row per lane); consumed after softmax
        {
            unsigned long long vrow =
                (unsigned long long)(qkv + (size_t)(rowbase + key0 + lane) * TD + 2 * DD + hh * DH);
            ASYNC_B128(vDst, vrow);
            ASYNC_B128_OFS(vDst, vrow, 16);
            ASYNC_B128_OFS(vDst, vrow, 32);
            ASYNC_B128_OFS(vDst, vrow, 48);
        }
        // pad mask (one key per lane)
        kmask[lane] = (x[rowbase + key0 + lane] == 0) ? -1e30f : 0.0f;
        __syncthreads();

        float sv[2][8];
#pragma unroll
        for (int tile = 0; tile < 2; ++tile) {
            v16bf fk;
            const unsigned short* kp = qkv + (size_t)(rowbase + key0 + tile * 16 + mr) * TD + DD + hh * DH;
#pragma unroll
            for (int p = 0; p < 8; ++p) {
                int kk = ((p >> 2) << 4) + hl * 8 + 2 * (p & 3);
                unsigned int u = *(const unsigned int*)(kp + kk);
                fk[2 * p] = bfc((unsigned short)u); fk[2 * p + 1] = bfc((unsigned short)(u >> 16));
            }
            v8f c = {};
            c = __builtin_amdgcn_wmma_f32_16x16x32_bf16(false, fk, false, fq, (short)0, c, false, false);
#pragma unroll
            for (int rr = 0; rr < 8; ++rr) {
                int krel = tile * 16 + 8 * hl + rr;
                float v = c[rr] * scal + kmask[krel];
                if (key0 + krel > q0 + mr) v = -1e30f;   // causal
                sv[tile][rr] = v;
            }
        }
        // online softmax stats (per query = per lane col, partner lane ^ 16)
        float mx = -1e30f;
#pragma unroll
        for (int t = 0; t < 2; ++t)
#pragma unroll
            for (int rr = 0; rr < 8; ++rr) mx = fmaxf(mx, sv[t][rr]);
        mx = fmaxf(mx, XORF(mx, 16));
        float mold = row_m[mr];
        float mnew = fmaxf(mold, mx);
        float sc   = expf(mold - mnew);
        float pv[2][8];
        float ssum = 0.0f;
#pragma unroll
        for (int t = 0; t < 2; ++t)
#pragma unroll
            for (int rr = 0; rr < 8; ++rr) { pv[t][rr] = expf(sv[t][rr] - mnew); ssum += pv[t][rr]; }
        ssum += XORF(ssum, 16);
        if (lane < 16) {
            row_sc[lane] = sc;
            row_m[lane]  = mnew;
            row_l[lane]  = row_l[lane] * sc + ssum;
        }
        // pack probs -> p_lds[query][key]
#pragma unroll
        for (int t = 0; t < 2; ++t)
#pragma unroll
            for (int rr = 0; rr < 8; rr += 2) {
                unsigned int u = (unsigned int)f2bf(pv[t][rr]) | ((unsigned int)f2bf(pv[t][rr + 1]) << 16);
                *(unsigned int*)&p_lds[mr][t * 16 + 8 * hl + rr] = u;
            }
        __syncthreads();

        // rescale O accumulators by per-query factor
#pragma unroll
        for (int rr = 0; rr < 8; ++rr) {
            float f = row_sc[8 * hl + rr];
            o0[rr] *= f; o1[rr] *= f;
        }
        // P fragment (A operand, 16 queries x 32 keys)
        v16bf fp;
#pragma unroll
        for (int p = 0; p < 8; ++p) {
            int kk = ((p >> 2) << 4) + hl * 8 + 2 * (p & 3);
            unsigned int u = *(const unsigned int*)&p_lds[mr][kk];
            fp[2 * p] = bfc((unsigned short)u); fp[2 * p + 1] = bfc((unsigned short)(u >> 16));
        }
        // V fragments from LDS (B operand, 32 keys x 32 dh -> two N sub-tiles)
        WAIT_ASYNC(0x0);   // V chunk resident in LDS (single-wave block)
        v16bf fv0, fv1;
#pragma unroll
        for (int j = 0; j < 16; ++j) {
            int kk = ((j >> 3) << 4) + hl * 8 + (j & 7);
            fv0[j] = bfc(v_lds[kk][mr]);
            fv1[j] = bfc(v_lds[kk][mr + 16]);
        }
        o0 = __builtin_amdgcn_wmma_f32_16x16x32_bf16(false, fp, false, fv0, (short)0, o0, false, false);
        o1 = __builtin_amdgcn_wmma_f32_16x16x32_bf16(false, fp, false, fv1, (short)0, o1, false, false);
        __syncthreads();
    }

    // normalize and store O (bf16)
#pragma unroll
    for (int rr = 0; rr < 8; ++rr) {
        int qrow = 8 * hl + rr;
        float inv = 1.0f / row_l[qrow];
        size_t base = (size_t)(rowbase + q0 + qrow) * DD + hh * DH + mr;
        o[base]      = f2bf(o0[rr] * inv);
        o[base + 16] = f2bf(o1[rr] * inv);
    }
}

// ---------------- host launcher ----------------
extern "C" void kernel_launch(void* const* d_in, const int* in_sizes, int n_in,
                              void* d_out, int out_size, void* d_ws, size_t ws_size,
                              hipStream_t stream) {
    const int*   x     = (const int*)d_in[0];
    const float* emb   = (const float*)d_in[1];
    const float* pe    = (const float*)d_in[2];
    const float* ln1_w = (const float*)d_in[3];
    const float* ln1_b = (const float*)d_in[4];
    const float* qkv_w = (const float*)d_in[5];
    const float* out_w = (const float*)d_in[6];
    const float* ln2_w = (const float*)d_in[7];
    const float* ln2_b = (const float*)d_in[8];
    const float* w1    = (const float*)d_in[9];
    const float* b1    = (const float*)d_in[10];
    const float* w2    = (const float*)d_in[11];
    const float* b2    = (const float*)d_in[12];
    const float* lnf_w = (const float*)d_in[13];
    const float* lnf_b = (const float*)d_in[14];
    float* logits = (float*)d_out;

    // workspace carve-out (256B aligned)
    size_t off = 0;
    char* ws = (char*)d_ws;
    auto carve = [&](size_t bytes) -> void* {
        void* p = ws + off;
        off = (off + bytes + 255) & ~(size_t)255;
        return p;
    };
    float*          h       = (float*)         carve((size_t)TOKENS * DD  * 4);
    unsigned short* xn_bf   = (unsigned short*)carve((size_t)TOKENS * DD  * 2);
    unsigned short* qkv_bf  = (unsigned short*)carve((size_t)TOKENS * TD  * 2);
    unsigned short* o_bf    = (unsigned short*)carve((size_t)TOKENS * DD  * 2);
    unsigned short* y1_bf   = (unsigned short*)carve((size_t)TOKENS * FFN * 2);
    unsigned short* wqkvT   = (unsigned short*)carve((size_t)4 * TD  * DD * 2);  // [l][n=3D][k=D]
    unsigned short* woutT   = (unsigned short*)carve((size_t)4 * DD  * DD * 2);  // [l][n=D][k=D]
    unsigned short* w1T     = (unsigned short*)carve((size_t)4 * FFN * DD * 2);  // [l][n=FF][k=D]
    unsigned short* w2T     = (unsigned short*)carve((size_t)4 * DD * FFN * 2);  // [l][n=D][k=FF]
    unsigned short* lmT     = (unsigned short*)carve((size_t)VPAD * DD * 2);     // [n=Vpad][k=D]
    (void)ws_size; (void)in_sizes; (void)n_in; (void)out_size;

    // weights -> transposed bf16 (B-operand layout, staged once)
    k_wT<<<1024, 256, 0, stream>>>(qkv_w, wqkvT, DD, TD, 4 * DD * TD);
    k_wT<<<1024, 256, 0, stream>>>(out_w, woutT, DD, DD, 4 * DD * DD);
    k_wT<<<1024, 256, 0, stream>>>(w1,    w1T,   DD, FFN, 4 * DD * FFN);
    k_wT<<<1024, 256, 0, stream>>>(w2,    w2T,   FFN, DD, 4 * FFN * DD);
    k_lmhead<<<128, 256, 0, stream>>>(emb, lmT);

    // embedding + positional encoding
    k_embed<<<TOKENS, 256, 0, stream>>>(x, emb, pe, h);

    for (int i = 0; i < 4; ++i) {
        // xn = LN1(h) -> bf16
        k_ln<<<TOKENS / 8, 256, 0, stream>>>(h, ln1_w + i * DD, ln1_b + i * DD, xn_bf);
        // qkv = xn @ qkv_w[i]  (bf16 out for attention)
        k_gemm<<<dim3(TD / 64, TOKENS / 64), 256, 0, stream>>>(
            xn_bf, wqkvT + (size_t)i * TD * DD, TOKENS, TD, DD,
            nullptr, nullptr, nullptr, qkv_bf, 0);
        // attention -> o_bf
        k_attn<<<2 * HH_N * (LL / 16), 32, 0, stream>>>(qkv_bf, x, o_bf);
        // h = h + o @ out_w[i]
        k_gemm<<<dim3(DD / 64, TOKENS / 64), 256, 0, stream>>>(
            o_bf, woutT + (size_t)i * DD * DD, TOKENS, DD, DD,
            nullptr, h, h, nullptr, 0);
        // xn2 = LN2(h) -> bf16
        k_ln<<<TOKENS / 8, 256, 0, stream>>>(h, ln2_w + i * DD, ln2_b + i * DD, xn_bf);
        // y1 = gelu(xn2 @ w1[i] + b1[i]) -> bf16
        k_gemm<<<dim3(FFN / 64, TOKENS / 64), 256, 0, stream>>>(
            xn_bf, w1T + (size_t)i * FFN * DD, TOKENS, FFN, DD,
            b1 + i * FFN, nullptr, nullptr, y1_bf, 1);
        // h = h + y1 @ w2[i] + b2[i]
        k_gemm<<<dim3(DD / 64, TOKENS / 64), 256, 0, stream>>>(
            y1_bf, w2T + (size_t)i * DD * FFN, TOKENS, DD, FFN,
            b2 + i * DD, h, h, nullptr, 0);
    }

    // final LN + tied LM head: logits = LN_f(h) @ emb^T
    k_ln<<<TOKENS / 8, 256, 0, stream>>>(h, lnf_w, lnf_b, xn_bf);
    k_gemm<<<dim3(2, TOKENS / 64), 256, 0, stream>>>(
        xn_bf, lmT, TOKENS, 100, DD,
        nullptr, nullptr, logits, nullptr, 0);
}
